// VQAutoEncoder_26439818674351
// MI455X (gfx1250) — matrix operand order repaired and checked by reading, not compiled, not run on hardware
//
#include <hip/hip_runtime.h>
#include <math.h>

// ---------------------------------------------------------------------------
// Types for CDNA5 WMMA (gfx1250, wave32)
// ---------------------------------------------------------------------------
typedef __bf16 bf16_t;
typedef __bf16 v16bf __attribute__((ext_vector_type(16)));
typedef float  v8f   __attribute__((ext_vector_type(8)));

#define WAVE 32

// ---------------------------------------------------------------------------
// Implicit-GEMM conv / GEMM kernel:  out[M,N] = A[M,K] * B(K,N) (+bias, +res)
//   A: pre-packed bf16, row-major [Mr x Kr]  (Mr mult of 128, Kr mult of 32)
//   B: built on the fly from f32 source (im2col for conv, direct for 1x1,
//      transposed read for attention's v^T operand)
//
// Block: 128 threads = 4 waves. Block tile 128(M) x 64(N), K step 32.
// Each wave owns two 16-row M-slices (wv*16 and wv*16+64) and all 64 N cols
// -> 8 WMMA per K-step per wave.
//
// Data movement tuned for the ISA fragment layouts:
//   * A fragment = two contiguous 16B chunks of the packed row -> two
//     global_load_b128 per lane, no LDS staging.
//   * B tile is staged into LDS in fragment-ready order Bst[nTile][lane][16]
//     so each lane reads its fragment with two ds_load_b128 (48B padded
//     stride -> bank-spread).
// ---------------------------------------------------------------------------
struct GemmP {
    const float*  src;       // B source (f32)
    const bf16_t* A;         // packed A operand
    const float*  bias;      // [OC] or nullptr
    const float*  residual;  // same layout as out, or nullptr
    float*        out;
    int C, H, W;             // B source logical dims (conv input)
    int OC;                  // true M
    int OH, OW;              // N = OH*OW
    int k, stride, padT, padL;
    int Kdim, Kr;
    int bTrans;              // B[k][n] = src[n*srcLd + k]
    int srcLd;
    int storeT;              // out[n*ldOut + m] instead of out[m*ldOut + n]
    int ldOut;
};

#define BM 128
#define BN 64
#define BK 32

__launch_bounds__(128)
__global__ void vq_gemm(GemmP p)
{
    // fragment-ready B tile: [n-tile 0..3][lane 0..31][16 elems + 8 pad]
    __shared__ __align__(16) bf16_t Bst[4][32][24];

    const int tid  = threadIdx.x;
    const int lane = tid & (WAVE - 1);
    const int wv   = tid >> 5;                 // 0..3
    const int hh   = lane >> 4;                // 0 or 1 (half-wave)
    const int mBase = blockIdx.y * BM;
    const int nBase = blockIdx.x * BN;
    const int N   = p.OH * p.OW;
    const int kk2 = p.k * p.k;

    // A rows for this lane's two M-tiles (packed buffer has Mr>=mBase+128 rows)
    const int mRow0 = mBase + (wv << 4) + (lane & 15);
    const bf16_t* Arow0 = p.A + (size_t)mRow0 * p.Kr;
    const bf16_t* Arow1 = Arow0 + (size_t)64 * p.Kr;

    // B staging task: this thread gathers a 16-deep K column for one N column
    const int cCol = tid & 63;                 // 0..63  (n-local)
    const int half = tid >> 6;                 // 0..1   (k half: rows 0-15 / 16-31)
    const int nCol = nBase + cCol;
    bf16_t* dstB = &Bst[cCol >> 4][(half << 4) | (cCol & 15)][0];

    int ohC = 0, owC = 0;
    if (nCol < N) { ohC = nCol / p.OW; owC = nCol - ohC * p.OW; }

    v8f acc[8] = {};

    for (int kb = 0; kb < p.Kr; kb += BK) {
        // ---- gather 16 B elements (k = kb + half*16 + e) for column nCol ----
        union { bf16_t e[16]; uint4 q[2]; } tmp;
        const int kOff = kb + (half << 4);
        if (nCol < N) {
            if (p.bTrans) {
                const float* sp = p.src + (size_t)nCol * p.srcLd + kOff;
                #pragma unroll
                for (int e = 0; e < 16; ++e)
                    tmp.e[e] = (bf16_t)((kOff + e) < p.Kdim ? sp[e] : 0.0f);
            } else {
                #pragma unroll
                for (int e = 0; e < 16; ++e) {
                    int kkk = kOff + e;
                    float v = 0.0f;
                    if (kkk < p.Kdim) {
                        int cc  = kkk / kk2;
                        int rem = kkk - cc * kk2;
                        int dy  = rem / p.k;
                        int dx  = rem - dy * p.k;
                        int ih  = ohC * p.stride + dy - p.padT;
                        int iw  = owC * p.stride + dx - p.padL;
                        if (ih >= 0 && ih < p.H && iw >= 0 && iw < p.W)
                            v = p.src[((size_t)cc * p.H + ih) * p.W + iw];
                    }
                    tmp.e[e] = (bf16_t)v;
                }
            }
        } else {
            #pragma unroll
            for (int e = 0; e < 16; ++e) tmp.e[e] = (bf16_t)0.0f;
        }
        ((uint4*)dstB)[0] = tmp.q[0];          // two ds_store_b128
        ((uint4*)dstB)[1] = tmp.q[1];
        __syncthreads();

        // ---- A fragments: two 16B contiguous chunks each (contiguous layout:
        //      lanes 0-15 hold K 0-15; lanes 16-31 hold K 16-31) ----
        union { v16bf v; uint4 q[2]; } a0, a1;
        a0.q[0] = *(const uint4*)(Arow0 + kb + (hh << 4));
        a0.q[1] = *(const uint4*)(Arow0 + kb + 8 + (hh << 4));
        a1.q[0] = *(const uint4*)(Arow1 + kb + (hh << 4));
        a1.q[1] = *(const uint4*)(Arow1 + kb + 8 + (hh << 4));

        // ---- 4 B fragments, 8 WMMA ----
        #pragma unroll
        for (int t = 0; t < 4; ++t) {
            union { v16bf v; uint4 q[2]; } bfg;
            const uint4* bp = (const uint4*)&Bst[t][lane][0];
            bfg.q[0] = bp[0];                   // two ds_load_b128
            bfg.q[1] = bp[1];
            acc[t] = __builtin_amdgcn_wmma_f32_16x16x32_bf16(
                         false, a0.v, false, bfg.v, (short)0, acc[t], false, false);
            acc[4 + t] = __builtin_amdgcn_wmma_f32_16x16x32_bf16(
                         false, a1.v, false, bfg.v, (short)0, acc[4 + t], false, false);
        }
        __syncthreads();
    }

    // ---- epilogue: C/D layout  M = v + 8*(lane>=16), N = lane&15 ----
    #pragma unroll
    for (int hm = 0; hm < 2; ++hm) {
        #pragma unroll
        for (int t = 0; t < 4; ++t) {
            union { v8f v; float e[8]; } cf; cf.v = acc[hm * 4 + t];
            #pragma unroll
            for (int vv = 0; vv < 8; ++vv) {
                int mg = mBase + hm * 64 + (wv << 4) + (hh << 3) + vv;
                int ng = nBase + (t << 4) + (lane & 15);
                if (mg < p.OC && ng < N) {
                    float val = cf.e[vv];
                    if (p.bias) val += p.bias[mg];
                    size_t off = p.storeT ? ((size_t)ng * p.ldOut + mg)
                                          : ((size_t)mg * p.ldOut + ng);
                    if (p.residual) val += p.residual[off];
                    p.out[off] = val;
                }
            }
        }
    }
}

// ---------------------------------------------------------------------------
// Pack f32 -> bf16 A operand  [Mr x Kr], zero-padded; optional transpose.
//   normal:    dst[m,k] = src[m*srcLd + k]
//   transpose: dst[m,k] = src[k*srcLd + m]
// ---------------------------------------------------------------------------
__global__ void vq_pack(const float* src, bf16_t* dst, int M, int Kdim,
                        int Mr, int Kr, int transpose, int srcLd)
{
    size_t total = (size_t)Mr * Kr;
    for (size_t i = (size_t)blockIdx.x * blockDim.x + threadIdx.x;
         i < total; i += (size_t)gridDim.x * blockDim.x) {
        int m  = (int)(i / Kr);
        int kk = (int)(i - (size_t)m * Kr);
        float v = 0.0f;
        if (m < M && kk < Kdim)
            v = transpose ? src[(size_t)kk * srcLd + m]
                          : src[(size_t)m * srcLd + kk];
        dst[i] = (bf16_t)v;
    }
}

// ---------------------------------------------------------------------------
// GroupNorm (32 groups): two-pass
// ---------------------------------------------------------------------------
__global__ void vq_gn_reduce(const float* x, int cpg, int HW, float* mu, float* var)
{
    const int g = blockIdx.x;
    const float* xp = x + (size_t)g * cpg * HW;
    const size_t n = (size_t)cpg * HW;
    float s = 0.f, s2 = 0.f;
    for (size_t i = threadIdx.x; i < n; i += blockDim.x) {
        float v = xp[i]; s += v; s2 += v * v;
    }
    __shared__ float sh[256], sh2[256];
    sh[threadIdx.x] = s; sh2[threadIdx.x] = s2;
    __syncthreads();
    for (int o = 128; o > 0; o >>= 1) {
        if (threadIdx.x < o) { sh[threadIdx.x] += sh[threadIdx.x + o];
                               sh2[threadIdx.x] += sh2[threadIdx.x + o]; }
        __syncthreads();
    }
    if (threadIdx.x == 0) {
        float m = sh[0] / (float)n;
        mu[g]  = m;
        var[g] = sh2[0] / (float)n - m * m;
    }
}

__global__ void vq_gn_apply(const float* x, float* y, const float* gamma,
                            const float* beta, const float* mu, const float* var,
                            int C, int HW, int cpg, int doSwish)
{
    size_t total = (size_t)C * HW;
    for (size_t i = (size_t)blockIdx.x * blockDim.x + threadIdx.x;
         i < total; i += (size_t)gridDim.x * blockDim.x) {
        int c = (int)(i / HW);
        int g = c / cpg;
        float t = (x[i] - mu[g]) * rsqrtf(var[g] + 1e-6f) * gamma[c] + beta[c];
        if (doSwish) t = t / (1.0f + __expf(-t));
        y[i] = t;
    }
}

// ---------------------------------------------------------------------------
// Row softmax with pre-scale: w[row,:] = softmax(scale * w[row,:])
// ---------------------------------------------------------------------------
__global__ void vq_softmax(float* w, int N, float scale)
{
    float* r = w + (size_t)blockIdx.x * N;
    __shared__ float red[256];
    float mx = -3.0e38f;
    for (int i = threadIdx.x; i < N; i += blockDim.x) mx = fmaxf(mx, r[i]);
    red[threadIdx.x] = mx; __syncthreads();
    for (int o = 128; o > 0; o >>= 1) {
        if (threadIdx.x < o) red[threadIdx.x] = fmaxf(red[threadIdx.x], red[threadIdx.x + o]);
        __syncthreads();
    }
    mx = red[0]; __syncthreads();
    float s = 0.f;
    for (int i = threadIdx.x; i < N; i += blockDim.x) {
        float e = __expf((r[i] - mx) * scale);
        r[i] = e; s += e;
    }
    red[threadIdx.x] = s; __syncthreads();
    for (int o = 128; o > 0; o >>= 1) {
        if (threadIdx.x < o) red[threadIdx.x] += red[threadIdx.x + o];
        __syncthreads();
    }
    float inv = 1.0f / red[0];
    for (int i = threadIdx.x; i < N; i += blockDim.x) r[i] *= inv;
}

// ---------------------------------------------------------------------------
// VQ helpers
// ---------------------------------------------------------------------------
__global__ void vq_rownorm(const float* cb, float* out, int NE, int D)
{
    int e = blockIdx.x * blockDim.x + threadIdx.x;
    if (e < NE) {
        float s = 0.f;
        for (int d = 0; d < D; ++d) { float v = cb[(size_t)e * D + d]; s += v * v; }
        out[e] = s;
    }
}

__global__ void vq_argmin(const float* dots, const float* ce2, int* idx, int NE, int NP)
{
    const int p = blockIdx.x;
    float best = 3.0e38f; int bi = 0;
    for (int e = threadIdx.x; e < NE; e += blockDim.x) {
        float d = ce2[e] - 2.0f * dots[(size_t)e * NP + p];
        if (d < best || (d == best && e < bi)) { best = d; bi = e; }
    }
    __shared__ float sv[256]; __shared__ int si[256];
    sv[threadIdx.x] = best; si[threadIdx.x] = bi;
    __syncthreads();
    for (int o = 128; o > 0; o >>= 1) {
        if (threadIdx.x < o) {
            if (sv[threadIdx.x + o] < sv[threadIdx.x] ||
                (sv[threadIdx.x + o] == sv[threadIdx.x] && si[threadIdx.x + o] < si[threadIdx.x])) {
                sv[threadIdx.x] = sv[threadIdx.x + o];
                si[threadIdx.x] = si[threadIdx.x + o];
            }
        }
        __syncthreads();
    }
    if (threadIdx.x == 0) idx[p] = si[0];
}

__global__ void vq_gather(const float* cb, const int* idx, float* zq, int C, int NP)
{
    size_t total = (size_t)C * NP;
    for (size_t i = (size_t)blockIdx.x * blockDim.x + threadIdx.x;
         i < total; i += (size_t)gridDim.x * blockDim.x) {
        int c = (int)(i / NP);
        int p = (int)(i - (size_t)c * NP);
        zq[i] = cb[(size_t)idx[p] * C + c];
    }
}

__global__ void vq_loss(const float* z, const float* zq, float* outLoss,
                        size_t n, float coeff)
{
    float s = 0.f;
    for (size_t i = (size_t)blockIdx.x * blockDim.x + threadIdx.x;
         i < n; i += (size_t)gridDim.x * blockDim.x) {
        float d = zq[i] - z[i]; s += d * d;
    }
    __shared__ float sh[256];
    sh[threadIdx.x] = s; __syncthreads();
    for (int o = 128; o > 0; o >>= 1) {
        if (threadIdx.x < o) sh[threadIdx.x] += sh[threadIdx.x + o];
        __syncthreads();
    }
    if (threadIdx.x == 0) atomicAdd(outLoss, sh[0] * coeff);
}

__global__ void vq_upsample2(const float* x, float* y, int C, int H, int W)
{
    const int H2 = 2 * H, W2 = 2 * W;
    size_t total = (size_t)C * H2 * W2;
    for (size_t i = (size_t)blockIdx.x * blockDim.x + threadIdx.x;
         i < total; i += (size_t)gridDim.x * blockDim.x) {
        int c = (int)(i / ((size_t)H2 * W2));
        int r = (int)(i - (size_t)c * H2 * W2);
        int oh = r / W2, ow = r - oh * W2;
        y[i] = x[((size_t)c * H + (oh >> 1)) * W + (ow >> 1)];
    }
}

// ===========================================================================
// Host-side orchestration
// ===========================================================================
struct ConvW { const float* b; const float* w; };
struct GNW   { const float* b; const float* g; };
struct ResW  { ConvW c1, c2, nin; bool hasNin; GNW n1, n2; };
struct AttnW { ConvW k; GNW norm; ConvW proj, q, v; };

static inline int alignN(int v, int a) { return (v + a - 1) & ~(a - 1); }

static void convL(hipStream_t s, const float* x, int C, int H, int W,
                  const ConvW& cv, int OC, int k, int stride, int padT, int padL,
                  int OH, int OW, float* out, const float* residual, bf16_t* pk)
{
    const int Kdim = C * k * k;
    const int Kr = alignN(Kdim, 32);
    const int Mr = alignN(OC, BM);
    vq_pack<<<2048, 256, 0, s>>>(cv.w, pk, OC, Kdim, Mr, Kr, 0, Kdim);
    GemmP g{ x, pk, cv.b, residual, out, C, H, W, OC, OH, OW,
             k, stride, padT, padL, Kdim, Kr, 0, 0, 0, OH * OW };
    dim3 gd((OH * OW + BN - 1) / BN, Mr / BM);
    vq_gemm<<<gd, 128, 0, s>>>(g);
}

static void gemmG(hipStream_t s, const bf16_t* A, int M, int Kdim,
                  const float* Bsrc, int N, float* out, int ldOut,
                  int storeT, int bTrans, int srcLd,
                  const float* bias, const float* residual)
{
    const int Kr = alignN(Kdim, 32);
    const int Mr = alignN(M, BM);
    GemmP g{ Bsrc, A, bias, residual, out, Kdim, 1, N, M, 1, N,
             1, 1, 0, 0, Kdim, Kr, bTrans, srcLd, storeT, ldOut };
    dim3 gd((N + BN - 1) / BN, Mr / BM);
    vq_gemm<<<gd, 128, 0, s>>>(g);
}

static void gnorm(hipStream_t s, const float* x, int C, int HW, const GNW& p,
                  bool swish, float* y, float* stats)
{
    int cpg = C / 32;
    vq_gn_reduce<<<32, 256, 0, s>>>(x, cpg, HW, stats, stats + 32);
    vq_gn_apply<<<2048, 256, 0, s>>>(x, y, p.g, p.b, stats, stats + 32,
                                     C, HW, cpg, swish ? 1 : 0);
}

static void resB(hipStream_t s, const float* x, int C, int H, int W,
                 const ResW& p, int OC, float* out,
                 float* t0, float* t1, float* t2, bf16_t* pk, float* stats)
{
    gnorm(s, x, C, H * W, p.n1, true, t0, stats);
    convL(s, t0, C, H, W, p.c1, OC, 3, 1, 1, 1, H, W, t1, nullptr, pk);
    gnorm(s, t1, OC, H * W, p.n2, true, t0, stats);
    const float* res = x;
    if (p.hasNin) {
        convL(s, x, C, H, W, p.nin, OC, 1, 1, 0, 0, H, W, t2, nullptr, pk);
        res = t2;
    }
    convL(s, t0, OC, H, W, p.c2, OC, 3, 1, 1, 1, H, W, out, res, pk);
}

// Attention at 512ch, 64x64 (HW=4096)
static void attnB(hipStream_t s, const float* x, const AttnW& p, float* out,
                  float* hn, float* q, float* kk, float* v,
                  float* wbig, bf16_t* pk, float* stats)
{
    const int C = 512, HW = 4096;
    gnorm(s, x, C, HW, p.norm, false, hn, stats);
    convL(s, hn, C, 64, 64, p.q, C, 1, 1, 0, 0, 64, 64, q,  nullptr, pk);
    convL(s, hn, C, 64, 64, p.k, C, 1, 1, 0, 0, 64, 64, kk, nullptr, pk);
    convL(s, hn, C, 64, 64, p.v, C, 1, 1, 0, 0, 64, 64, v,  nullptr, pk);
    // w_ = q^T k : A = q^T [HW x C], B = k [C x HW]
    vq_pack<<<2048, 256, 0, s>>>(q, pk, HW, C, alignN(HW, BM), alignN(C, 32), 1, HW);
    gemmG(s, pk, HW, C, kk, HW, wbig, HW, 0, 0, 0, nullptr, nullptr);
    vq_softmax<<<HW, 256, 0, s>>>(wbig, HW, rsqrtf((float)C));
    // h^T = w_ [HW x HW] * v^T [HW x C], stored transposed -> h_ [C x HW]
    vq_pack<<<2048, 256, 0, s>>>(wbig, pk, HW, HW, alignN(HW, BM), alignN(HW, 32), 0, HW);
    float* h_ = hn;  // hn no longer needed
    gemmG(s, pk, HW, HW, v, C, h_, HW, 1, 1, HW, nullptr, nullptr);
    convL(s, h_, C, 64, 64, p.proj, C, 1, 1, 0, 0, 64, 64, out, x, pk);
}

// ---------------------------------------------------------------------------
extern "C" void kernel_launch(void* const* d_in, const int* in_sizes, int n_in,
                              void* d_out, int out_size, void* d_ws, size_t ws_size,
                              hipStream_t stream)
{
    (void)in_sizes; (void)n_in; (void)out_size; (void)ws_size;

    // ---- workspace layout (bump offsets, 64MB slots) ----
    const size_t MB = 1ull << 20;
    char* w = (char*)d_ws;
    float* A0 = (float*)(w + 0 * 64 * MB);
    float* A1 = (float*)(w + 1 * 64 * MB);
    float* A2 = (float*)(w + 2 * 64 * MB);
    float* A3 = (float*)(w + 3 * 64 * MB);
    float* A4 = (float*)(w + 4 * 64 * MB);
    float* A5 = (float*)(w + 5 * 64 * MB);
    bf16_t* PK  = (bf16_t*)(w + 6 * 64 * MB);   // 64MB packed A operand
    float*  BIG = (float*)(w + 7 * 64 * MB);    // 128MB: attention w_ / codebook dots
    float*  SM  = (float*)(w + 576 * MB);       // gn stats (mu[32], var[32])
    float*  CE2 = SM + 1024;                    // 8192 codebook norms
    int*    IDX = (int*)(w + 577 * MB);         // 4096 indices

    // ---- walk d_in: "x" first, then params leaves (jax tree sorted keys) ----
    int ti = 0;
    auto F = [&]() { return (const float*)d_in[ti++]; };
    auto getConv = [&]() { ConvW c; c.b = F(); c.w = F(); return c; };
    auto getGN   = [&]() { GNW g; g.b = F(); g.g = F(); return g; };
    auto getRes  = [&](bool nin) {
        ResW r; r.c1 = getConv(); r.c2 = getConv();
        r.hasNin = nin; if (nin) r.nin = getConv();
        r.n1 = getGN(); r.n2 = getGN(); return r;
    };
    auto getAttn = [&]() {
        AttnW a; a.k = getConv(); a.norm = getGN();
        a.proj = getConv(); a.q = getConv(); a.v = getConv(); return a;
    };

    const float* X = F();                       // x [1,3,256,256]
    const float* codebook = F();                // codebook [8192,256]
    // --- dec: conv_in, conv_out, mid1, mid2, mid_attn, norm_out, up ---
    ConvW dec_conv_in  = getConv();
    ConvW dec_conv_out = getConv();
    ResW  dec_mid1 = getRes(false);
    ResW  dec_mid2 = getRes(false);
    AttnW dec_attn = getAttn();
    GNW   dec_norm_out = getGN();
    ResW  u0b[3]; for (int j = 0; j < 3; ++j) u0b[j] = getRes(false);
    ConvW u0up = getConv();
    ResW  u1b[3]; u1b[0] = getRes(true); u1b[1] = getRes(false); u1b[2] = getRes(false);
    ConvW u1up = getConv();
    ResW  u2b[3]; u2b[0] = getRes(true); u2b[1] = getRes(false); u2b[2] = getRes(false);
    // --- enc: conv_in, conv_out, down, mid1, mid2, mid_attn, norm_out ---
    ConvW enc_conv_in  = getConv();
    ConvW enc_conv_out = getConv();
    ResW e0b0 = getRes(false), e0b1 = getRes(false); ConvW e0d = getConv();
    ResW e1b0 = getRes(true),  e1b1 = getRes(false); ConvW e1d = getConv();
    ResW e2b0 = getRes(true),  e2b1 = getRes(false);
    ResW enc_mid1 = getRes(false), enc_mid2 = getRes(false);
    AttnW enc_attn = getAttn();
    GNW enc_norm_out = getGN();
    ConvW post_quant = getConv();
    ConvW quant      = getConv();

    float* O = (float*)d_out;
    hipMemsetAsync(O + 196608, 0, sizeof(float), stream);   // emb_loss accumulator

    // ======================= ENCODER =======================
    convL(stream, X, 3, 256, 256, enc_conv_in, 128, 3, 1, 1, 1, 256, 256, A0, nullptr, PK);
    resB(stream, A0, 128, 256, 256, e0b0, 128, A1, A2, A3, A4, PK, SM);
    resB(stream, A1, 128, 256, 256, e0b1, 128, A0, A2, A3, A4, PK, SM);
    // downsample: torch pad (0,1,0,1) + stride2 conv, pad0
    convL(stream, A0, 128, 256, 256, e0d, 128, 3, 2, 0, 0, 128, 128, A1, nullptr, PK);
    resB(stream, A1, 128, 128, 128, e1b0, 256, A0, A2, A3, A4, PK, SM);
    resB(stream, A0, 256, 128, 128, e1b1, 256, A1, A2, A3, A4, PK, SM);
    convL(stream, A1, 256, 128, 128, e1d, 256, 3, 2, 0, 0, 64, 64, A0, nullptr, PK);
    resB(stream, A0, 256, 64, 64, e2b0, 512, A1, A2, A3, A4, PK, SM);
    resB(stream, A1, 512, 64, 64, e2b1, 512, A0, A2, A3, A4, PK, SM);
    resB(stream, A0, 512, 64, 64, enc_mid1, 512, A1, A2, A3, A4, PK, SM);
    attnB(stream, A1, enc_attn, A0, A2, A3, A4, A5, BIG, PK, SM);
    resB(stream, A0, 512, 64, 64, enc_mid2, 512, A1, A2, A3, A4, PK, SM);
    gnorm(stream, A1, 512, 4096, enc_norm_out, true, A0, SM);
    convL(stream, A0, 512, 64, 64, enc_conv_out, 256, 3, 1, 1, 1, 64, 64, A1, nullptr, PK);
    convL(stream, A1, 256, 64, 64, quant, 256, 1, 1, 0, 0, 64, 64, A2, nullptr, PK); // z = A2

    // ======================= QUANTIZE =======================
    // dots[e,p] = codebook[e,:] . z[:,p]  via WMMA GEMM (M=8192,K=256,N=4096)
    vq_pack<<<2048, 256, 0, stream>>>(codebook, PK, 8192, 256, 8192, 256, 0, 256);
    gemmG(stream, PK, 8192, 256, A2, 4096, BIG, 4096, 0, 0, 0, nullptr, nullptr);
    vq_rownorm<<<32, 256, 0, stream>>>(codebook, CE2, 8192, 256);
    vq_argmin<<<4096, 256, 0, stream>>>(BIG, CE2, IDX, 8192, 4096);
    vq_gather<<<2048, 256, 0, stream>>>(codebook, IDX, A3, 256, 4096);   // z_q = A3
    vq_loss<<<1024, 256, 0, stream>>>(A2, A3, O + 196608, (size_t)256 * 4096,
                                      1.25f / (256.0f * 4096.0f));
    convL(stream, A3, 256, 64, 64, post_quant, 256, 1, 1, 0, 0, 64, 64, A0, nullptr, PK);

    // ======================= DECODER =======================
    convL(stream, A0, 256, 64, 64, dec_conv_in, 512, 3, 1, 1, 1, 64, 64, A1, nullptr, PK);
    resB(stream, A1, 512, 64, 64, dec_mid1, 512, A0, A2, A3, A4, PK, SM);
    attnB(stream, A0, dec_attn, A1, A2, A3, A4, A5, BIG, PK, SM);
    resB(stream, A1, 512, 64, 64, dec_mid2, 512, A0, A2, A3, A4, PK, SM);
    // up level 0 (512ch @ 64^2, then upsample -> 128^2)
    resB(stream, A0, 512, 64, 64, u0b[0], 512, A1, A2, A3, A4, PK, SM);
    resB(stream, A1, 512, 64, 64, u0b[1], 512, A0, A2, A3, A4, PK, SM);
    resB(stream, A0, 512, 64, 64, u0b[2], 512, A1, A2, A3, A4, PK, SM);
    vq_upsample2<<<4096, 256, 0, stream>>>(A1, A0, 512, 64, 64);
    convL(stream, A0, 512, 128, 128, u0up, 512, 3, 1, 1, 1, 128, 128, A1, nullptr, PK);
    // up level 1 (512->256 @ 128^2, upsample -> 256^2)
    resB(stream, A1, 512, 128, 128, u1b[0], 256, A0, A2, A3, A4, PK, SM);
    resB(stream, A0, 256, 128, 128, u1b[1], 256, A1, A2, A3, A4, PK, SM);
    resB(stream, A1, 256, 128, 128, u1b[2], 256, A0, A2, A3, A4, PK, SM);
    vq_upsample2<<<4096, 256, 0, stream>>>(A0, A1, 256, 128, 128);
    convL(stream, A1, 256, 256, 256, u1up, 256, 3, 1, 1, 1, 256, 256, A0, nullptr, PK);
    // up level 2 (256->128 @ 256^2)
    resB(stream, A0, 256, 256, 256, u2b[0], 128, A1, A2, A3, A4, PK, SM);
    resB(stream, A1, 128, 256, 256, u2b[1], 128, A0, A2, A3, A4, PK, SM);
    resB(stream, A0, 128, 256, 256, u2b[2], 128, A1, A2, A3, A4, PK, SM);
    gnorm(stream, A1, 128, 65536, dec_norm_out, true, A0, SM);
    // final conv writes straight into d_out [3,256,256]
    convL(stream, A0, 128, 256, 256, dec_conv_out, 3, 3, 1, 1, 1, 256, 256, O, nullptr, PK);
}
